// LSTMN_86354612453910
// MI455X (gfx1250) — matrix operand order repaired and checked by reading, not compiled, not run on hardware
//
#include <hip/hip_runtime.h>
#include <hip/hip_bf16.h>
#include <cstdint>

// ---------------- types ----------------
typedef __bf16 bf16_t;
typedef __bf16 v16bf __attribute__((ext_vector_type(16)));
typedef __bf16 v8bf  __attribute__((ext_vector_type(8)));
typedef float  v8f   __attribute__((ext_vector_type(8)));

// ---------------- optional CDNA5 async global->LDS path (probe-gated) ----------------
#if defined(__has_builtin)
#  if __has_builtin(__builtin_amdgcn_global_load_async_to_lds_b128) && \
      __has_builtin(__builtin_amdgcn_s_wait_asynccnt)
#    define USE_ASYNC_LDS 1
#  endif
#endif
#ifndef USE_ASYNC_LDS
#  define USE_ASYNC_LDS 0
#endif

#if USE_ASYNC_LDS
typedef int v4i_t __attribute__((ext_vector_type(4)));
typedef __attribute__((address_space(1))) v4i_t* gv4_ptr;   // global int4*
typedef __attribute__((address_space(3))) v4i_t* lv4_ptr;   // LDS    int4*
__device__ __forceinline__ void async_cp16(void* dstLds, const void* srcGlobal) {
  // flat global address == AS1 address; flat LDS address truncates to LDS offset (ISA 10.2)
  gv4_ptr g = (gv4_ptr)(unsigned long long)(uintptr_t)srcGlobal;
  lv4_ptr l = (lv4_ptr)(unsigned int)(uintptr_t)dstLds;
  __builtin_amdgcn_global_load_async_to_lds_b128(g, l, 0, 0);
}
__device__ __forceinline__ void async_join() { __builtin_amdgcn_s_wait_asynccnt(0); }
#endif

// ---------------- problem constants ----------------
constexpr int T_TOTAL = 2048;
constexpr int TAPE    = 30;
constexpr int HID     = 512;
constexpr int FOURH   = 2048;
constexpr int NWG     = 32;   // scan workgroups (one 16-wide hidden slice each)

// ---------------- workspace layout (bytes) ----------------
constexpr size_t OFF_CNT   = 0;                        // 256 (barrier counter)
constexpr size_t OFF_WIH   = 256;                      // 2*2048*512*2 = 4194304
constexpr size_t OFF_WHH   = OFF_WIH  + 4194304;       // 4194304
constexpr size_t OFF_WX    = OFF_WHH  + 4194304;       // 2*512*512*2 = 1048576
constexpr size_t OFF_XBF   = OFF_WX   + 1048576;       // 2048*512*2  = 2097152
constexpr size_t OFF_XW0   = OFF_XBF  + 2097152;       // 2048*2048*4 = 16777216
constexpr size_t OFF_XA    = OFF_XW0  + 16777216;      // 2*2048*512*4 = 8388608
constexpr size_t OFF_DWH   = OFF_XA   + 8388608;       // 4096
constexpr size_t OFF_DWHT  = OFF_DWH  + 4096;          // 4096
constexpr size_t OFF_TAPEH = OFF_DWHT + 4096;          // 2*30*512*4 = 122880
constexpr size_t OFF_TAPEC = OFF_TAPEH + 122880;       // 122880
constexpr size_t OFF_H0BF  = OFF_TAPEC + 122880;       // 32*512*2 = 32768
constexpr size_t OFF_G1    = OFF_H0BF  + 32768;        // 32*2048*4 = 262144
constexpr size_t OFF_BC    = OFF_G1    + 262144;       // broadcast buffers

// ---------------- helpers ----------------
__device__ __forceinline__ float sigf(float x) { return 1.0f / (1.0f + expf(-x)); }

__device__ __forceinline__ v8f zero8() {
  v8f z;
#pragma unroll
  for (int i = 0; i < 8; ++i) z[i] = 0.0f;
  return z;
}

__device__ __forceinline__ v16bf load_a_frag(const bf16_t* rowPtr, int kb, int hi) {
  // A fragment in contiguous k order (B uses the same, not the ISA table order): lanes 0-15 K=0..15, lanes 16-31 K=16..31
  v8bf x0 = *(const v8bf*)(rowPtr + kb + hi * 16);
  v8bf x1 = *(const v8bf*)(rowPtr + kb + hi * 16 + 8);
  v16bf a;
#pragma unroll
  for (int j = 0; j < 8; ++j) { a[j] = x0[j]; a[j + 8] = x1[j]; }
  return a;
}

__device__ __forceinline__ v16bf load_b_vec(const bf16_t* vecLds, int kb, int r, int hi) {
  // B 32x16: lane n<16 holds K=0..15 of col n, lane n+16 holds K=16..31. Vector in col 0 only.
  v16bf b;
#pragma unroll
  for (int j = 0; j < 16; ++j) b[j] = (bf16_t)0.0f;
  if (r == 0) {
    v8bf y0 = *(const v8bf*)(vecLds + kb + hi * 16);
    v8bf y1 = *(const v8bf*)(vecLds + kb + hi * 16 + 8);
#pragma unroll
    for (int j = 0; j < 8; ++j) { b[j] = y0[j]; b[j + 8] = y1[j]; }
  }
  return b;
}

__device__ __forceinline__ v8f wmma_bf16(v16bf a, v16bf b, v8f c) {
  return __builtin_amdgcn_wmma_f32_16x16x32_bf16(false, a, false, b, (short)0, c, false, false);
}

// ---------------- elementwise / setup kernels ----------------
__global__ void cvt_bf16_kernel(const float* __restrict__ s, bf16_t* __restrict__ d, int n) {
  int i = blockIdx.x * 256 + threadIdx.x;
  if (i < n) d[i] = (bf16_t)s[i];
}

__global__ void init_misc_kernel(const float* __restrict__ wh, const float* __restrict__ wht,
                                 float* __restrict__ dwh, float* __restrict__ dwht,
                                 unsigned* __restrict__ cnt) {
  int i = blockIdx.x * 256 + threadIdx.x;
  if (i == 0) *cnt = 0u;
  if (i < 2 * HID) {
    int l = i >> 9, h = i & 511;
    size_t base = (size_t)l * HID * HID + (size_t)h * (HID + 1);
    dwh[i]  = wh[base];
    dwht[i] = wht[base];
  }
}

__global__ void zero_tail_kernel(float* __restrict__ out) {
  int i = blockIdx.x * 256 + threadIdx.x;   // 30*512 elements
  if (i < TAPE * HID) out[(size_t)(T_TOTAL - TAPE) * HID + i] = 0.0f;
}

// ---------------- WMMA GEMM: C[n][m] = sum_k A[m][k]*B[n][k] (+bias0[m]+bias1[m]) ----------------
__global__ void __launch_bounds__(256)
gemm_bf16_kernel(const bf16_t* __restrict__ A, const bf16_t* __restrict__ B,
                 float* __restrict__ C, int M, int N, int K,
                 const float* __restrict__ bias0, const float* __restrict__ bias1) {
  const int lane = threadIdx.x & 31;
  const int wv   = threadIdx.x >> 5;
  const int tile = blockIdx.x * 8 + wv;
  const int mt = M >> 4, nt = N >> 4;
  if (tile >= mt * nt) return;
  const int m0 = (tile / nt) * 16;
  const int n0 = (tile % nt) * 16;
  const int r = lane & 15, hi = lane >> 4;

  const bf16_t* aRow = A + (size_t)(m0 + r) * K;
  const bf16_t* bRow = B + (size_t)(n0 + r) * K + hi * 16;
  v8f acc = zero8();
  for (int kb = 0; kb < K; kb += 32) {
    if ((kb & 127) == 0) {
      __builtin_prefetch(aRow + kb + 256, 0, 1);
      __builtin_prefetch(bRow + kb + 256, 0, 1);
    }
    v16bf af = load_a_frag(aRow, kb, hi);
    v8bf y0 = *(const v8bf*)(bRow + kb);
    v8bf y1 = *(const v8bf*)(bRow + kb + 8);
    v16bf bfrag;
#pragma unroll
    for (int j = 0; j < 8; ++j) { bfrag[j] = y0[j]; bfrag[j + 8] = y1[j]; }
    acc = wmma_bf16(af, bfrag, acc);
  }
  // D layout: VGPR rr -> M = hi*8+rr ; N = lane&15
  float* cp = C + (size_t)(n0 + r) * M + m0 + hi * 8;
#pragma unroll
  for (int rr = 0; rr < 8; ++rr) {
    float v = acc[rr];
    if (bias0) { int mrow = m0 + hi * 8 + rr; v += bias0[mrow] + bias1[mrow]; }
    cp[rr] = v;
  }
}

// ---------------- tape init (phase 1: independent steps from zero state) ----------------
__global__ void tape1_kernel(const float* __restrict__ XW0, bf16_t* __restrict__ H0bf,
                             float* __restrict__ tapeHg, float* __restrict__ tapeCg) {
  int i = blockIdx.x * 256 + threadIdx.x;   // 32*512
  if (i >= 32 * HID) return;
  int t = i >> 9, h = i & 511;
  const float* g = XW0 + (size_t)t * FOURH;       // biases already folded in
  float gi = g[h], gg = g[1024 + h], go = g[1536 + h];
  float c = sigf(gi) * tanhf(gg);                  // c_prev = 0 -> f-gate term vanishes
  float hh = sigf(go) * tanhf(c);
  H0bf[(size_t)t * HID + h] = (bf16_t)hh;
  if (t < TAPE) {
    tapeHg[(size_t)t * HID + h] = hh;              // layer 0 slots [0..29]
    tapeCg[(size_t)t * HID + h] = c;
  }
}

__global__ void tape2_kernel(const float* __restrict__ G1,
                             float* __restrict__ tapeHg, float* __restrict__ tapeCg) {
  int i = blockIdx.x * 256 + threadIdx.x;   // 30*512
  if (i >= TAPE * HID) return;
  int t = i >> 9, h = i & 511;
  const float* g = G1 + (size_t)t * FOURH;        // biases folded in by GEMM epilogue
  float gi = g[h], gg = g[1024 + h], go = g[1536 + h];
  float c = sigf(gi) * tanhf(gg);
  float hh = sigf(go) * tanhf(c);
  tapeHg[(size_t)(TAPE + t) * HID + h] = hh;       // layer 1 slots [30..59]
  tapeCg[(size_t)(TAPE + t) * HID + h] = c;
}

// ---------------- grid barrier ----------------
__device__ __forceinline__ void grid_barrier(unsigned* cnt, unsigned* gen) {
  __syncthreads();
  if (threadIdx.x == 0) {
    __threadfence();
    atomicAdd(cnt, 1u);
    unsigned target = (++(*gen)) * (unsigned)NWG;
    while (atomicAdd(cnt, 0u) < target) __builtin_amdgcn_s_sleep(2);
    __threadfence();
  }
  __syncthreads();
}

// ---------------- WMMA matvec slice: 4 gate tiles x 16 rows, K=512 per matrix ----------------
__device__ __forceinline__ void matvec_stage(const bf16_t* __restrict__ W1,
                                             const bf16_t* __restrict__ W2,
                                             const bf16_t* v1Lds, const bf16_t* v2Lds,
                                             float* sPart, int m0) {
  const int lane = threadIdx.x & 31;
  const int wv   = threadIdx.x >> 5;
  const int r = lane & 15, hi = lane >> 4;
#pragma unroll
  for (int q = 0; q < 4; ++q) {
    v8f acc = zero8();
    const size_t rowOff = (size_t)(q * 512 + m0 + r) * HID;
#pragma unroll
    for (int kk = 0; kk < 2; ++kk) {
      const int kb = wv * 64 + kk * 32;
      acc = wmma_bf16(load_a_frag(W1 + rowOff, kb, hi), load_b_vec(v1Lds, kb, r, hi), acc);
      if (W2)
        acc = wmma_bf16(load_a_frag(W2 + rowOff, kb, hi), load_b_vec(v2Lds, kb, r, hi), acc);
    }
    if (r == 0) {            // column N=0 lives in lanes 0 (M=0..7) and 16 (M=8..15)
#pragma unroll
      for (int rr = 0; rr < 8; ++rr) sPart[(q * 16 + hi * 8 + rr) * 8 + wv] = acc[rr];
    }
  }
}

// ---------------- persistent sequential scan ----------------
__global__ void __launch_bounds__(256, 1)
scan_kernel(const float* __restrict__ XW0, const float* __restrict__ XAv,
            const float* __restrict__ tapeHg, const float* __restrict__ tapeCg,
            const bf16_t* __restrict__ WihBf, const bf16_t* __restrict__ WhhBf,
            const float* __restrict__ dwhG, const float* __restrict__ dwhtG,
            const float* __restrict__ attnV, const float* __restrict__ bih,
            const float* __restrict__ bhh,
            float* __restrict__ bH0f, float* __restrict__ bC0f,
            float* __restrict__ bH1f, float* __restrict__ bC1f,
            bf16_t* __restrict__ bH0bf,
            unsigned* __restrict__ cnt, float* __restrict__ out) {
  __shared__ __align__(16) float sTapeH[2 * TAPE * HID];   // 122880 B
  __shared__ __align__(16) float sTapeC[2 * TAPE * HID];   // 122880 B
  __shared__ float sPrevH[2 * HID], sPrevC[2 * HID];
  __shared__ __align__(16) bf16_t sPrevHbf[2 * HID];
  __shared__ __align__(16) bf16_t sH0bf[HID];
  __shared__ __align__(16) float sXA[2 * HID];
  __shared__ float sDwh[2 * HID], sDwht[2 * HID], sV[2 * HID];
  __shared__ float sScore[64], sAlpha[64];
  __shared__ float sPart[512], sG[64];

  const int tid  = threadIdx.x;
  const int lane = tid & 31;
  const int wv   = tid >> 5;
  const int m0   = blockIdx.x * 16;
  unsigned gen = 0;

#if USE_ASYNC_LDS
  for (int i = tid; i < (2 * TAPE * HID) / 4; i += 256) {  // 16B chunks straight into LDS
    async_cp16(&sTapeH[i * 4], &tapeHg[i * 4]);
    async_cp16(&sTapeC[i * 4], &tapeCg[i * 4]);
  }
#else
  for (int i = tid; i < 2 * TAPE * HID; i += 256) { sTapeH[i] = tapeHg[i]; sTapeC[i] = tapeCg[i]; }
#endif
  for (int i = tid; i < 2 * HID; i += 256) {
    sDwh[i] = dwhG[i]; sDwht[i] = dwhtG[i]; sV[i] = attnV[i];
    sPrevH[i] = 0.0f; sPrevC[i] = 0.0f; sPrevHbf[i] = (bf16_t)0.0f;
  }
#if USE_ASYNC_LDS
  async_join();
#endif
  __syncthreads();

  for (int s = 0; s < T_TOTAL - TAPE; ++s) {
    const int t = s + TAPE;

    // ---- P1: attention over the tape (redundant per workgroup, all in LDS) ----
#if USE_ASYNC_LDS
    {
      const int i = tid * 4;                     // 1024 floats total, 16B per thread
      const int l = i >> 9;
      async_cp16(&sXA[i], &XAv[(size_t)l * T_TOTAL * HID + (size_t)t * HID + (i & 511)]);
      async_join();
    }
#else
    for (int i = tid; i < 2 * HID; i += 256)
      sXA[i] = XAv[(size_t)(i >> 9) * T_TOTAL * HID + (size_t)t * HID + (i & 511)];
#endif
    __syncthreads();

    for (int p = wv; p < 2 * TAPE; p += 8) {
      const int l = p / TAPE, k = p % TAPE;
      const float* th = &sTapeH[(l * TAPE + k) * HID];
      float sacc = 0.0f;
      for (int h = lane; h < HID; h += 32) {
        float a = tanhf(sDwh[l * HID + h] * th[h] + sXA[l * HID + h] +
                        sDwht[l * HID + h] * sPrevH[l * HID + h]);
        sacc += a * sV[l * HID + h];
      }
#pragma unroll
      for (int msk = 16; msk >= 1; msk >>= 1) sacc += __shfl_xor(sacc, msk, 32);
      if (lane == 0) sScore[l * 32 + k] = sacc;
    }
    __syncthreads();

    if (wv < 2) {
      const int l = wv;
      float sc = (lane < TAPE) ? sScore[l * 32 + lane] : -3.0e38f;
      float mx = sc;
#pragma unroll
      for (int msk = 16; msk >= 1; msk >>= 1) mx = fmaxf(mx, __shfl_xor(mx, msk, 32));
      float e = (lane < TAPE) ? expf(sc - mx) : 0.0f;
      float sum = e;
#pragma unroll
      for (int msk = 16; msk >= 1; msk >>= 1) sum += __shfl_xor(sum, msk, 32);
      if (lane < TAPE) sAlpha[l * 32 + lane] = e / sum;
    }
    __syncthreads();

    for (int i = tid; i < 2 * HID; i += 256) {
      const int l = i >> 9, h = i & 511;
      const float* tH = &sTapeH[l * TAPE * HID + h];
      const float* tC = &sTapeC[l * TAPE * HID + h];
      const float* al = &sAlpha[l * 32];
      float ah = 0.0f, ac = 0.0f;
#pragma unroll 6
      for (int k = 0; k < TAPE; ++k) { float a = al[k]; ah += a * tH[k * HID]; ac += a * tC[k * HID]; }
      sPrevH[i] = ah; sPrevC[i] = ac; sPrevHbf[i] = (bf16_t)ah;
    }
    __syncthreads();

    // ---- P2: layer-0 recurrent matvec for this WG's 16-wide slice ----
    matvec_stage(WhhBf, nullptr, sPrevHbf, nullptr, sPart, m0);
    __syncthreads();
    if (tid < 64) {
      float g = 0.0f;
#pragma unroll
      for (int w = 0; w < 8; ++w) g += sPart[tid * 8 + w];
      const int q = tid >> 4, ml = tid & 15;
      g += XW0[(size_t)t * FOURH + q * 512 + m0 + ml];   // includes W_ih[0]@x + biases
      sG[tid] = g;
    }
    __syncthreads();
    if (tid < 16) {
      const int m = m0 + tid;
      float gi = sG[tid], gf = sG[16 + tid], gg = sG[32 + tid], go = sG[48 + tid];
      float c = sigf(gf) * sPrevC[m] + sigf(gi) * tanhf(gg);
      float h = sigf(go) * tanhf(c);
      bH0f[m] = h; bC0f[m] = c; bH0bf[m] = (bf16_t)h;
    }
    grid_barrier(cnt, &gen);

    // ---- P3: gather h0, append layer-0 tape slot, layer-1 matvecs ----
    const int slot = s % TAPE;
#if USE_ASYNC_LDS
    for (int c = tid; c < 320; c += 256) {               // 1KB + 2KB + 2KB in 16B chunks
      if (c < 64)       async_cp16(&sH0bf[c * 8], &bH0bf[c * 8]);
      else if (c < 192) { int i = (c - 64) * 4;  async_cp16(&sTapeH[slot * HID + i], &bH0f[i]); }
      else              { int i = (c - 192) * 4; async_cp16(&sTapeC[slot * HID + i], &bC0f[i]); }
    }
    async_join();
#else
    for (int i = tid; i < HID; i += 256) {
      sH0bf[i] = bH0bf[i];
      sTapeH[slot * HID + i] = bH0f[i];
      sTapeC[slot * HID + i] = bC0f[i];
    }
#endif
    __syncthreads();
    matvec_stage(WihBf + (size_t)FOURH * HID, WhhBf + (size_t)FOURH * HID,
                 sH0bf, sPrevHbf + HID, sPart, m0);
    __syncthreads();
    if (tid < 64) {
      float g = 0.0f;
#pragma unroll
      for (int w = 0; w < 8; ++w) g += sPart[tid * 8 + w];
      const int q = tid >> 4, ml = tid & 15;
      const int row = q * 512 + m0 + ml;
      g += bih[FOURH + row] + bhh[FOURH + row];
      sG[tid] = g;
    }
    __syncthreads();
    if (tid < 16) {
      const int m = m0 + tid;
      float gi = sG[tid], gf = sG[16 + tid], gg = sG[32 + tid], go = sG[48 + tid];
      float c = sigf(gf) * sPrevC[HID + m] + sigf(gi) * tanhf(gg);
      float h = sigf(go) * tanhf(c);
      out[(size_t)s * HID + m] = h;
      bH1f[m] = h; bC1f[m] = c;
    }
    grid_barrier(cnt, &gen);

    // ---- P4: append layer-1 tape slot ----
#if USE_ASYNC_LDS
    {
      const int c = tid;                                  // 2KB + 2KB in 16B chunks
      if (c < 128) async_cp16(&sTapeH[(TAPE + slot) * HID + c * 4], &bH1f[c * 4]);
      else         async_cp16(&sTapeC[(TAPE + slot) * HID + (c - 128) * 4], &bC1f[(c - 128) * 4]);
      async_join();
    }
#else
    for (int i = tid; i < HID; i += 256) {
      sTapeH[(TAPE + slot) * HID + i] = bH1f[i];
      sTapeC[(TAPE + slot) * HID + i] = bC1f[i];
    }
#endif
    __syncthreads();
  }
}

// ---------------- launch ----------------
extern "C" void kernel_launch(void* const* d_in, const int* in_sizes, int n_in,
                              void* d_out, int out_size, void* d_ws, size_t ws_size,
                              hipStream_t stream) {
  (void)in_sizes; (void)n_in; (void)out_size; (void)ws_size;
  const float* xs   = (const float*)d_in[0];
  const float* Wih  = (const float*)d_in[1];
  const float* Whh  = (const float*)d_in[2];
  const float* bih  = (const float*)d_in[3];
  const float* bhh  = (const float*)d_in[4];
  const float* awh  = (const float*)d_in[5];
  const float* awx  = (const float*)d_in[6];
  const float* awht = (const float*)d_in[7];
  const float* av   = (const float*)d_in[8];
  float* out = (float*)d_out;

  char* ws = (char*)d_ws;
  unsigned* cnt   = (unsigned*)(ws + OFF_CNT);
  bf16_t* WihBf   = (bf16_t*)(ws + OFF_WIH);
  bf16_t* WhhBf   = (bf16_t*)(ws + OFF_WHH);
  bf16_t* WxBf    = (bf16_t*)(ws + OFF_WX);
  bf16_t* Xbf     = (bf16_t*)(ws + OFF_XBF);
  float*  XW0     = (float*)(ws + OFF_XW0);
  float*  XAv     = (float*)(ws + OFF_XA);
  float*  dwh     = (float*)(ws + OFF_DWH);
  float*  dwht    = (float*)(ws + OFF_DWHT);
  float*  tapeHg  = (float*)(ws + OFF_TAPEH);
  float*  tapeCg  = (float*)(ws + OFF_TAPEC);
  bf16_t* H0bf    = (bf16_t*)(ws + OFF_H0BF);
  float*  G1      = (float*)(ws + OFF_G1);
  float*  bH0f    = (float*)(ws + OFF_BC);
  float*  bC0f    = (float*)(ws + OFF_BC + 2048);
  float*  bH1f    = (float*)(ws + OFF_BC + 4096);
  float*  bC1f    = (float*)(ws + OFF_BC + 6144);
  bf16_t* bH0bf   = (bf16_t*)(ws + OFF_BC + 8192);

  // 1) fp32 -> bf16 conversions
  cvt_bf16_kernel<<<(2097152 + 255) / 256, 256, 0, stream>>>(Wih, WihBf, 2097152);
  cvt_bf16_kernel<<<(2097152 + 255) / 256, 256, 0, stream>>>(Whh, WhhBf, 2097152);
  cvt_bf16_kernel<<<(524288  + 255) / 256, 256, 0, stream>>>(awx, WxBf, 524288);
  cvt_bf16_kernel<<<(1048576 + 255) / 256, 256, 0, stream>>>(xs,  Xbf,  1048576);

  // 2) diagonals + barrier-counter reset (required every launch for determinism)
  init_misc_kernel<<<4, 256, 0, stream>>>(awh, awht, dwh, dwht, cnt);

  // 3) big WMMA GEMMs over all timesteps (independent of the recurrence)
  //    XW0[t][j] = W_ih[0]@x[t] + b_ih[0] + b_hh[0]
  gemm_bf16_kernel<<<2048, 256, 0, stream>>>(WihBf, Xbf, XW0, FOURH, T_TOTAL, HID, bih, bhh);
  //    XA[l][t][h] = attn_wx[l]@x[t]
  gemm_bf16_kernel<<<512, 256, 0, stream>>>(WxBf, Xbf, XAv, HID, T_TOTAL, HID, nullptr, nullptr);
  gemm_bf16_kernel<<<512, 256, 0, stream>>>(WxBf + (size_t)HID * HID, Xbf,
                                            XAv + (size_t)T_TOTAL * HID, HID, T_TOTAL, HID,
                                            nullptr, nullptr);

  // 4) phase-1 tape init (30 independent steps from zero state)
  tape1_kernel<<<64, 256, 0, stream>>>(XW0, H0bf, tapeHg, tapeCg);
  gemm_bf16_kernel<<<32, 256, 0, stream>>>(WihBf + (size_t)FOURH * HID, H0bf, G1,
                                           FOURH, 32, HID, bih + FOURH, bhh + FOURH);
  tape2_kernel<<<60, 256, 0, stream>>>(G1, tapeHg, tapeCg);

  // 5) output tail (rows 2018..2047 are zero in the reference)
  zero_tail_kernel<<<60, 256, 0, stream>>>(out);

  // 6) persistent sequential scan: 32 WGs, LDS-resident tape, 2 grid barriers/step
  scan_kernel<<<NWG, 256, 0, stream>>>(XW0, XAv, tapeHg, tapeCg, WihBf, WhhBf,
                                       dwh, dwht, av, bih, bhh,
                                       bH0f, bC0f, bH1f, bC1f, bH0bf, cnt, out);
}
